// IEModule_32444182954698
// MI455X (gfx1250) — matrix operand order repaired and checked by reading, not compiled, not run on hardware
//
#include <hip/hip_runtime.h>
#include <hip/hip_bf16.h>

typedef __attribute__((ext_vector_type(16))) _Float16     v16h;
typedef __attribute__((ext_vector_type(8)))  _Float16     v8h;
typedef __attribute__((ext_vector_type(8)))  float        v8f;
typedef __attribute__((ext_vector_type(4)))  float        v4f;
typedef __attribute__((ext_vector_type(4)))  unsigned int v4u;
typedef __attribute__((ext_vector_type(8)))  unsigned int v8u;

#define B_ 8
#define N_ 2048
#define C_ 128
#define D_ 32
#define NWAVE 4
#define PRS 40   // LDS p-tile row stride in halves (80B: 16B-aligned, bank-spread)

__device__ __forceinline__ v8f wmma_f16(v16h a, v16h b, v8f c) {
  // D = A(16x32 f16) * B(32x16 f16) + C(16x16 f32)
  return __builtin_amdgcn_wmma_f32_16x16x32_f16(false, a, false, b, (short)0, c, false, false);
}

__device__ __forceinline__ v16h combine16h(v4u a, v4u b) {
  v8u u = {a[0], a[1], a[2], a[3], b[0], b[1], b[2], b[3]};
  return __builtin_bit_cast(v16h, u);
}
// A-layout gather: 16 contiguous halves p[0..15]  (K runs base+0..15)
__device__ __forceinline__ v16h loadA16h(const _Float16* p) {
  v4u a = *(const v4u*)p;
  v4u b = *(const v4u*)(p + 8);
  return combine16h(a, b);
}
// B-layout gather: 16 contiguous halves
__device__ __forceinline__ v16h loadC16h(const _Float16* p) {
  v4u a = *(const v4u*)p;
  v4u b = *(const v4u*)(p + 8);
  return combine16h(a, b);
}

// ---------------------------------------------------------------------------
// Generic projection: Y[b,co,n] = sum_ci W[co,ci] * X[b,ci,n] (+bias)
// X: [B, Cin, N] f32. Writes f32 and/or f16; f16 optionally transposed
// ([b][n][Cout], used for q so WMMA operand loads are contiguous).
// ---------------------------------------------------------------------------
__global__ void proj_kernel(const float* __restrict__ X, const float* __restrict__ W,
                            const float* __restrict__ bias, float* __restrict__ Yf,
                            _Float16* __restrict__ Yh, int htrans, int Cin, int Cout) {
  __shared__ float xs[128 * 64];           // Cin (<=128) x 64-column tile
  const int NT = 64;
  const int b  = blockIdx.y;
  const int n0 = blockIdx.x * NT;
  const float* Xb = X + (size_t)b * Cin * N_;
  for (int t = threadIdx.x; t < Cin * NT; t += blockDim.x) {
    int ci = t / NT, n = t % NT;
    xs[t] = Xb[(size_t)ci * N_ + n0 + n];
  }
  __syncthreads();
  for (int o = threadIdx.x; o < Cout * NT; o += blockDim.x) {
    int co = o / NT, n = o % NT;
    float s = bias ? bias[co] : 0.f;
    const float* Wr = W + (size_t)co * Cin;
#pragma unroll 4
    for (int ci = 0; ci < Cin; ++ci) s = fmaf(Wr[ci], xs[ci * NT + n], s);
    if (Yf) Yf[(size_t)b * Cout * N_ + (size_t)co * N_ + n0 + n] = s;
    if (Yh) {
      if (htrans) Yh[((size_t)b * N_ + n0 + n) * D_ + co] = (_Float16)s;
      else        Yh[(size_t)b * Cout * N_ + (size_t)co * N_ + n0 + n] = (_Float16)s;
    }
  }
}

// ---------------------------------------------------------------------------
// Per-channel BN stats over (B,N): stats[2c]=mu, stats[2c+1]=rsqrt(var+eps)
// ---------------------------------------------------------------------------
__global__ void bn_stats_kernel(const float* __restrict__ pre, float* __restrict__ stats, int C) {
  __shared__ float s1[256], s2[256];
  const int c = blockIdx.x;
  const int M = B_ * N_;
  float a = 0.f, q = 0.f;
  for (int t = threadIdx.x; t < M; t += 256) {
    int b = t >> 11, n = t & (N_ - 1);
    float v = pre[(size_t)b * C * N_ + (size_t)c * N_ + n];
    a += v; q += v * v;
  }
  s1[threadIdx.x] = a; s2[threadIdx.x] = q;
  __syncthreads();
  for (int st = 128; st > 0; st >>= 1) {
    if (threadIdx.x < st) { s1[threadIdx.x] += s1[threadIdx.x + st]; s2[threadIdx.x] += s2[threadIdx.x + st]; }
    __syncthreads();
  }
  if (threadIdx.x == 0) {
    float mu  = s1[0] / (float)M;
    float var = s2[0] / (float)M - mu * mu;
    stats[2 * c]     = mu;
    stats[2 * c + 1] = rsqrtf(var + 1e-5f);
  }
}

__global__ void bn_relu_kernel(const float* __restrict__ pre, const float* __restrict__ stats,
                               const float* __restrict__ g, const float* __restrict__ bb,
                               float* __restrict__ out, int C) {
  size_t total = (size_t)B_ * C * N_;
  for (size_t i = (size_t)blockIdx.x * blockDim.x + threadIdx.x; i < total;
       i += (size_t)gridDim.x * blockDim.x) {
    int c = (int)((i / N_) % C);
    float y = g[c] * (pre[i] - stats[2 * c]) * stats[2 * c + 1] + bb[c];
    out[i] = y > 0.f ? y : 0.f;
  }
}

// h <- h + relu(bn(pre)); also scatter into d_out channel slice [blk*128,(blk+1)*128)
__global__ void bn_relu_res_kernel(const float* __restrict__ pre, const float* __restrict__ stats,
                                   const float* __restrict__ g, const float* __restrict__ bb,
                                   float* __restrict__ h, float* __restrict__ out, int blk) {
  size_t total = (size_t)B_ * C_ * N_;
  for (size_t i = (size_t)blockIdx.x * blockDim.x + threadIdx.x; i < total;
       i += (size_t)gridDim.x * blockDim.x) {
    int n = (int)(i % N_);
    int c = (int)((i / N_) % C_);
    int b = (int)(i / ((size_t)C_ * N_));
    float y = g[c] * (pre[i] - stats[2 * c]) * stats[2 * c + 1] + bb[c];
    y = y > 0.f ? y : 0.f;
    float hn = h[i] + y;
    h[i] = hn;
    out[(size_t)b * 512 * N_ + (size_t)(blk * 128 + c) * N_ + n] = hn;
  }
}

// ---------------------------------------------------------------------------
// Pass 1: softmax row stats of energy = q^T q (never materialized).
// Two sweeps: (1) max-only (pure v_max, no transcendentals),
//             (2) l += exp(e - m) (one exp per element, no rescale chain).
// Energy tiles recomputed by WMMA both times (K=32, operands L2-resident).
// ---------------------------------------------------------------------------
__global__ void attn_rowstats_kernel(const _Float16* __restrict__ qT,
                                     float* __restrict__ mrow, float* __restrict__ rlin) {
  const int b    = blockIdx.x;
  const int wave = threadIdx.x >> 5;
  const int lane = threadIdx.x & 31;
  const int l    = lane & 15;
  const int hi   = lane >> 4;
  const int i0   = (blockIdx.y * NWAVE + wave) * 16;
  const _Float16* qb = qT + (size_t)b * N_ * D_;

  // A tile: A[m,k=d] = qT[i0+m][d]; lane row m=l, K base = hi*16
  v16h a = loadA16h(qb + (size_t)(i0 + l) * D_ + hi * 16);
  v8f zero = {};

  // ---- sweep 1: row max ----
  float m8[8];
#pragma unroll
  for (int r = 0; r < 8; ++r) m8[r] = -3.0e38f;
  for (int j0 = 0; j0 < N_; j0 += 16) {
    v16h bq = loadC16h(qb + (size_t)(j0 + l) * D_ + hi * 16);
    v8f ec = wmma_f16(a, bq, zero);
#pragma unroll
    for (int r = 0; r < 8; ++r) m8[r] = fmaxf(m8[r], ec[r]);
  }
#pragma unroll
  for (int mask = 1; mask <= 8; mask <<= 1) {
#pragma unroll
    for (int r = 0; r < 8; ++r) m8[r] = fmaxf(m8[r], __shfl_xor(m8[r], mask, 32));
  }

  // ---- sweep 2: row sum of exp(e - m) ----
  float l8[8];
#pragma unroll
  for (int r = 0; r < 8; ++r) l8[r] = 0.f;
  for (int j0 = 0; j0 < N_; j0 += 16) {
    v16h bq = loadC16h(qb + (size_t)(j0 + l) * D_ + hi * 16);
    v8f ec = wmma_f16(a, bq, zero);
#pragma unroll
    for (int r = 0; r < 8; ++r) l8[r] += __expf(ec[r] - m8[r]);
  }
#pragma unroll
  for (int mask = 1; mask <= 8; mask <<= 1) {
#pragma unroll
    for (int r = 0; r < 8; ++r) l8[r] += __shfl_xor(l8[r], mask, 32);
  }

  if (l == 0) {
#pragma unroll
    for (int r = 0; r < 8; ++r) {
      int row = i0 + hi * 8 + r;
      mrow[b * N_ + row] = m8[r];
      rlin[b * N_ + row] = 1.f / l8[r];
    }
  }
}

// ---------------------------------------------------------------------------
// Pass 2: per 16-col tile j: recompute energy (WMMA), p = exp(e-m_i)/l_i,
// accumulate colsum_j and acc[c,j] += v[c,i]*p[i,j] (8 WMMAs, C=128),
// then t[c,j] = h[c,j] - acc[c,j]/(1e-9 + colsum_j).
// p repacked C-layout -> B-layout through LDS with b128 ds ops.
// ---------------------------------------------------------------------------
__global__ void attn_apply_kernel(const _Float16* __restrict__ qT, const _Float16* __restrict__ v,
                                  const float* __restrict__ mrow, const float* __restrict__ rlin,
                                  const float* __restrict__ h, float* __restrict__ tbuf) {
  __shared__ _Float16 psh[NWAVE * 16 * PRS];   // [wave][n(16)][k(32)+pad]
  const int b    = blockIdx.x;
  const int wave = threadIdx.x >> 5;
  const int lane = threadIdx.x & 31;
  const int l    = lane & 15;
  const int hi   = lane >> 4;
  const int j0   = (blockIdx.y * NWAVE + wave) * 16;
  const _Float16* qb = qT + (size_t)b * N_ * D_;
  const _Float16* vb = v + (size_t)b * C_ * N_;
  const float* mb = mrow + b * N_;
  const float* rb = rlin + b * N_;
  _Float16* prow = psh + (wave * 16 + l) * PRS;

  // fixed B operand for energy: columns j0..j0+15 of q  (qT rows, contiguous)
  v16h bq = loadC16h(qb + (size_t)(j0 + l) * D_ + hi * 16);

  v8f zero = {};
  v8f acc[8];
#pragma unroll
  for (int t = 0; t < 8; ++t) acc[t] = zero;
  float colp = 0.f;

  for (int i0 = 0; i0 < N_; i0 += 32) {
    __builtin_prefetch(qb + (size_t)(i0 + 32 + l) * D_, 0, 1);
#pragma unroll
    for (int h2 = 0; h2 < 2; ++h2) {
      // A tile: A[m,k=d] = qT[i0+h2*16+m][d]
      v16h a = loadA16h(qb + (size_t)(i0 + h2 * 16 + l) * D_ + hi * 16);
      v8f ec = wmma_f16(a, bq, zero);
      int rowbase = i0 + h2 * 16 + hi * 8;
      v4f m0 = *(const v4f*)(mb + rowbase);
      v4f m1 = *(const v4f*)(mb + rowbase + 4);
      v4f r0 = *(const v4f*)(rb + rowbase);
      v4f r1 = *(const v4f*)(rb + rowbase + 4);
      v8h pv;
#pragma unroll
      for (int r = 0; r < 8; ++r) {
        float mm = (r < 4) ? m0[r & 3] : m1[r & 3];
        float rr = (r < 4) ? r0[r & 3] : r1[r & 3];
        float p  = __expf(ec[r] - mm) * rr;
        colp += p;
        pv[r] = (_Float16)p;
      }
      // one b128 LDS store: p[n=l][k = h2*16 + hi*8 .. +7]
      *(v4u*)(prow + h2 * 16 + hi * 8) = __builtin_bit_cast(v4u, pv);
    }
    __builtin_amdgcn_wave_barrier();
    // B operand: B[k,n=l] = p[l][k base hi*16], 16 contiguous halves -> 2x ds b128
    v16h bp = loadC16h(prow + hi * 16);
    __builtin_amdgcn_wave_barrier();
#pragma unroll
    for (int ct = 0; ct < 8; ++ct) {
      // A tile: A[m=c_local,k=i_local] = v[ct*16+m][i0+k]; contiguous K runs
      const _Float16* vr = vb + (size_t)(ct * 16 + l) * N_ + i0 + hi * 16;
      acc[ct] = wmma_f16(loadA16h(vr), bp, acc[ct]);
    }
  }

  float cso  = __shfl_xor(colp, 16, 32);
  float invc = 1.f / (1e-9f + colp + cso);
#pragma unroll
  for (int ct = 0; ct < 8; ++ct) {
#pragma unroll
    for (int r = 0; r < 8; ++r) {
      int c = ct * 16 + hi * 8 + r;
      size_t idx = (size_t)b * C_ * N_ + (size_t)c * N_ + j0 + l;
      tbuf[idx] = h[idx] - acc[ct][r] * invc;
    }
  }
}

// ---------------------------------------------------------------------------
extern "C" void kernel_launch(void* const* d_in, const int* in_sizes, int n_in,
                              void* d_out, int out_size, void* d_ws, size_t ws_size,
                              hipStream_t stream) {
  (void)in_sizes; (void)n_in; (void)out_size; (void)ws_size;
  const float* x   = (const float*)d_in[0];
  const float* W1  = (const float*)d_in[1];
  const float* g1  = (const float*)d_in[2];
  const float* b1  = (const float*)d_in[3];
  const float* W2  = (const float*)d_in[4];
  const float* g2  = (const float*)d_in[5];
  const float* b2  = (const float*)d_in[6];
  const float* Wqk = (const float*)d_in[7];
  const float* Wv  = (const float*)d_in[8];
  const float* bv  = (const float*)d_in[9];
  const float* Wt  = (const float*)d_in[10];
  const float* bt  = (const float*)d_in[11];
  const float* gn  = (const float*)d_in[12];
  const float* bnb = (const float*)d_in[13];
  float* out = (float*)d_out;

  char* ws = (char*)d_ws;
  size_t off = 0;
  auto alloc = [&](size_t bytes) -> char* {
    char* p = ws + off;
    off += (bytes + 255) & ~(size_t)255;
    return p;
  };
  float*     pre   = (float*)alloc(sizeof(float) * B_ * C_ * N_);       // 8 MB
  float*     h64   = (float*)alloc(sizeof(float) * B_ * 64 * N_);       // 4 MB
  float*     h     = (float*)alloc(sizeof(float) * B_ * C_ * N_);       // 8 MB
  float*     tbuf  = (float*)alloc(sizeof(float) * B_ * C_ * N_);       // 8 MB
  _Float16*  qTh   = (_Float16*)alloc(sizeof(_Float16) * B_ * N_ * D_); // 1 MB
  _Float16*  vh    = (_Float16*)alloc(sizeof(_Float16) * B_ * C_ * N_); // 4 MB
  float*     mrow  = (float*)alloc(sizeof(float) * B_ * N_);
  float*     rlin  = (float*)alloc(sizeof(float) * B_ * N_);
  float*     stats = (float*)alloc(sizeof(float) * 2 * C_);

  dim3 pgrid(N_ / 64, B_);
  dim3 agrid(B_, N_ / (16 * NWAVE));

  // MLP stem: h = relu(bn(W2 @ relu(bn(W1 @ x))))
  proj_kernel<<<pgrid, 256, 0, stream>>>(x, W1, nullptr, pre, nullptr, 0, 64, 64);
  bn_stats_kernel<<<64, 256, 0, stream>>>(pre, stats, 64);
  bn_relu_kernel<<<1024, 256, 0, stream>>>(pre, stats, g1, b1, h64, 64);
  proj_kernel<<<pgrid, 256, 0, stream>>>(h64, W2, nullptr, pre, nullptr, 0, 64, 128);
  bn_stats_kernel<<<128, 256, 0, stream>>>(pre, stats, 128);
  bn_relu_kernel<<<2048, 256, 0, stream>>>(pre, stats, g2, b2, h, 128);

  for (int i = 0; i < 4; ++i) {
    proj_kernel<<<pgrid, 256, 0, stream>>>(h, Wqk + (size_t)i * 32 * 128, nullptr,
                                           nullptr, qTh, 1, 128, 32);
    proj_kernel<<<pgrid, 256, 0, stream>>>(h, Wv + (size_t)i * 128 * 128, bv + i * 128,
                                           nullptr, vh, 0, 128, 128);
    attn_rowstats_kernel<<<agrid, 32 * NWAVE, 0, stream>>>(qTh, mrow, rlin);
    attn_apply_kernel<<<agrid, 32 * NWAVE, 0, stream>>>(qTh, vh, mrow, rlin, h, tbuf);
    proj_kernel<<<pgrid, 256, 0, stream>>>(tbuf, Wt + (size_t)i * 128 * 128, bt + i * 128,
                                           pre, nullptr, 0, 128, 128);
    bn_stats_kernel<<<128, 256, 0, stream>>>(pre, stats, 128);
    bn_relu_res_kernel<<<2048, 256, 0, stream>>>(pre, stats, gn + i * 128, bnb + i * 128,
                                                 h, out, i);
  }
}
